// Critic_90194313216640
// MI455X (gfx1250) — matrix operand order repaired and checked by reading, not compiled, not run on hardware
//
#include <hip/hip_runtime.h>
#include <hip/hip_bf16.h>

#define NB 64    // N nodes
#define BB 128   // batch
#define HH 256   // hidden
#define DD 128   // out dim
#define KK 8     // neighbors
#define LDSK 40  // padded LDS k-stride in ushorts (80B, 16B-aligned)

typedef __attribute__((ext_vector_type(16))) __bf16 v16bf;
typedef __attribute__((ext_vector_type(8)))  float  v8f;
typedef unsigned int u32x4 __attribute__((ext_vector_type(4)));
typedef unsigned int u32x2 __attribute__((ext_vector_type(2)));
typedef float        f32x4 __attribute__((ext_vector_type(4)));

union FragU { u32x4 q[2]; v16bf v; };

// f32 -> bf16 (round-half-up; 2 VALU)
__device__ __forceinline__ unsigned short f2bf(float f) {
  return (unsigned short)((__float_as_uint(f) + 0x8000u) >> 16);
}

// pack two f32 -> packed bf16x2 (low = x, high = y)
__device__ __forceinline__ unsigned pack_bf16(float x, float y) {
#if __has_builtin(__builtin_amdgcn_cvt_pk_bf16_f32)
  auto r = __builtin_amdgcn_cvt_pk_bf16_f32(x, y);
  unsigned u;
  __builtin_memcpy(&u, &r, sizeof(u));
  return u;
#elif __has_builtin(__builtin_amdgcn_perm)
  // v_perm_b32 selects the two rounded high halves: 3 VALU per pair
  unsigned ux = __float_as_uint(x) + 0x8000u;
  unsigned uy = __float_as_uint(y) + 0x8000u;
  return __builtin_amdgcn_perm(uy, ux, 0x07060302u);
#else
  return (unsigned)f2bf(x) | ((unsigned)f2bf(y) << 16);
#endif
}

// A fragment (16x32 bf16): lane<16 holds K 0..7,16..23 ; lane>=16 holds K 8..15,24..31
__device__ __forceinline__ v16bf frag_a(const unsigned short* row, int lane) {
  const int kb = (lane < 16) ? 0 : 8;
  FragU f;
  f.q[0] = *(const u32x4*)(row + kb);
  f.q[1] = *(const u32x4*)(row + kb + 16);
  return f.v;
}
// B fragment (32x16 bf16): lane<16 holds K 0..7,16..23 ; lane>=16 holds K 8..15,24..31 (col = lane&15)
__device__ __forceinline__ v16bf frag_b(const unsigned short* row, int lane) {
  const int kb = (lane < 16) ? 0 : 8;
  FragU f;
  f.q[0] = *(const u32x4*)(row + kb);
  f.q[1] = *(const u32x4*)(row + kb + 16);
  return f.v;
}

// ---------------------------------------------------------------------------
// Kernel 1: H1 = relu(LN(X @ W1 + b1)) -> bf16 Hout (N,B,H)
// grid = (n=64, mb=2, path=2). Block = 64x256 tile; waves 2x4, tiles 2x4.
// Double-buffered LDS + register-pipelined staging (load / wmma / convert+store).
// ---------------------------------------------------------------------------
__global__ __launch_bounds__(256) void mlp1_ln_relu(
    const float* __restrict__ obs, const float* __restrict__ act,
    const float* __restrict__ V_W1, const float* __restrict__ V_b1,
    const float* __restrict__ V_g1, const float* __restrict__ V_be1,
    const float* __restrict__ A_W1, const float* __restrict__ A_b1,
    const float* __restrict__ A_g1, const float* __restrict__ A_be1,
    unsigned short* __restrict__ Hv, unsigned short* __restrict__ Ha)
{
  __shared__ __align__(16) unsigned short Xs[2][64][LDSK];   // 10 KB
  __shared__ __align__(16) unsigned short Ws[2][HH][LDSK];   // 40 KB  [o][k]
  __shared__ float red_s[64][4];
  __shared__ float red_q[64][4];
  __shared__ float s_mean[64];
  __shared__ float s_rstd[64];

  const int n    = blockIdx.x;
  const int r0   = blockIdx.y * 64;          // batch-row base
  const int path = blockIdx.z;               // 0 = V, 1 = A
  const float* W1  = path ? A_W1  : V_W1;
  const float* b1  = path ? A_b1  : V_b1;
  const float* g1  = path ? A_g1  : V_g1;
  const float* be1 = path ? A_be1 : V_be1;
  unsigned short* Hout = path ? Ha : Hv;
  const int KDIM = path ? 2*HH : HH;

  const int tid  = threadIdx.x;
  const int lane = tid & 31;
  const int w    = tid >> 5;
  const int wm   = w >> 2;      // 0..1 : 32-row band
  const int wn   = w & 3;       // 0..3 : 64-col band
  const int l15  = lane & 15;
  const int halfsel = lane >> 4;

  // staging thread mapping
  const int xb  = tid >> 3;            // X: row 0..31 base, 2 iters cover 64
  const int xkq = (tid & 7) << 2;      // X: k quad
  const int wog   = (tid & 63) << 2;   // W: 4 output columns
  const int wkbas = (tid >> 6) << 3;   // W: 8 consecutive k

  f32x4 xr[2], wr[8];

  auto stage_load = [&](int k0) {
    const float* xsrc = (k0 < HH) ? obs : act;
    const int koff = ((k0 < HH) ? k0 : k0 - HH) + xkq;
    #pragma unroll
    for (int j = 0; j < 2; ++j)
      xr[j] = *(const f32x4*)&xsrc[((size_t)(r0 + xb + j*32)*NB + n)*HH + koff];
    #pragma unroll
    for (int j = 0; j < 8; ++j)
      wr[j] = *(const f32x4*)&W1[((size_t)n*(size_t)KDIM + (k0 + wkbas + j))*HH + wog];
  };
  auto stage_store = [&](int buf) {
    #pragma unroll
    for (int j = 0; j < 2; ++j) {
      u32x2 p;
      p.x = pack_bf16(xr[j].x, xr[j].y);
      p.y = pack_bf16(xr[j].z, xr[j].w);
      *(u32x2*)&Xs[buf][xb + j*32][xkq] = p;
    }
    #pragma unroll
    for (int c = 0; c < 4; ++c) {
      u32x4 p;
      p[0] = pack_bf16(wr[0][c], wr[1][c]);
      p[1] = pack_bf16(wr[2][c], wr[3][c]);
      p[2] = pack_bf16(wr[4][c], wr[5][c]);
      p[3] = pack_bf16(wr[6][c], wr[7][c]);
      *(u32x4*)&Ws[buf][wog + c][wkbas] = p;
    }
  };

  v8f acc[2][4];
  #pragma unroll
  for (int a = 0; a < 2; ++a)
    #pragma unroll
    for (int c = 0; c < 4; ++c) {
      v8f z = {0.f,0.f,0.f,0.f,0.f,0.f,0.f,0.f};
      acc[a][c] = z;
    }

  const int nk = KDIM >> 5;
  stage_load(0);
  stage_store(0);
  __syncthreads();
  for (int kc = 0; kc < nk; ++kc) {
    const int cur = kc & 1;
    const bool more = (kc + 1 < nk);
    if (more) stage_load((kc + 1) << 5);      // global loads issue here ...
    v16bf af[2], bfr[4];
    #pragma unroll
    for (int tm = 0; tm < 2; ++tm)
      af[tm] = frag_a(&Xs[cur][wm*32 + tm*16 + l15][0], lane);
    #pragma unroll
    for (int tn = 0; tn < 4; ++tn)
      bfr[tn] = frag_b(&Ws[cur][wn*64 + tn*16 + l15][0], lane);
    #pragma unroll
    for (int tm = 0; tm < 2; ++tm)
      #pragma unroll
      for (int tn = 0; tn < 4; ++tn)
        acc[tm][tn] = __builtin_amdgcn_wmma_f32_16x16x32_bf16(
            false, af[tm], false, bfr[tn], (short)0, acc[tm][tn], false, false);
    if (more) stage_store(cur ^ 1);           // ... and are only awaited after wmma
    __syncthreads();
  }

  // + b1
  #pragma unroll
  for (int tn = 0; tn < 4; ++tn) {
    float bias = b1[(size_t)n*HH + wn*64 + tn*16 + l15];
    #pragma unroll
    for (int tm = 0; tm < 2; ++tm)
      #pragma unroll
      for (int v = 0; v < 8; ++v)
        acc[tm][tn][v] += bias;
  }

  // per-row sum / sumsq across the 4 column waves
  #pragma unroll
  for (int tm = 0; tm < 2; ++tm) {
    #pragma unroll
    for (int v = 0; v < 8; ++v) {
      float s = 0.f, q = 0.f;
      #pragma unroll
      for (int tn = 0; tn < 4; ++tn) {
        float a = acc[tm][tn][v];
        s += a; q += a * a;
      }
      #pragma unroll
      for (int off = 1; off < 16; off <<= 1) {
        s += __shfl_xor(s, off, 32);
        q += __shfl_xor(q, off, 32);
      }
      if (l15 == 0) {
        int row = wm*32 + tm*16 + v + (halfsel ? 8 : 0);
        red_s[row][wn] = s;
        red_q[row][wn] = q;
      }
    }
  }
  __syncthreads();
  if (tid < 64) {
    float s = red_s[tid][0] + red_s[tid][1] + red_s[tid][2] + red_s[tid][3];
    float q = red_q[tid][0] + red_q[tid][1] + red_q[tid][2] + red_q[tid][3];
    float mean = s * (1.f/256.f);
    float var  = q * (1.f/256.f) - mean*mean;
    s_mean[tid] = mean;
    s_rstd[tid] = rsqrtf(var + 1e-5f);
  }
  __syncthreads();

  // normalize, scale/shift, relu, emit bf16
  #pragma unroll
  for (int tn = 0; tn < 4; ++tn) {
    int col = wn*64 + tn*16 + l15;
    float gv = g1[col];
    float bv = be1[col];
    #pragma unroll
    for (int tm = 0; tm < 2; ++tm)
      #pragma unroll
      for (int v = 0; v < 8; ++v) {
        int row = wm*32 + tm*16 + v + (halfsel ? 8 : 0);
        float y = (acc[tm][tn][v] - s_mean[row]) * s_rstd[row] * gv + bv;
        y = fmaxf(y, 0.f);
        Hout[((size_t)n*BB + r0 + row)*HH + col] = f2bf(y);
      }
  }
}

// ---------------------------------------------------------------------------
// Kernel 2: Out = H @ W2 + b2  (per-n 128x256 @ 256x128), f32 out (N,B,D)
// grid = (n=64, mb=2, path=2). Block 64x128; waves 2x4, tiles 2x2.
// ---------------------------------------------------------------------------
__global__ __launch_bounds__(256) void mlp2_kernel(
    const unsigned short* __restrict__ Hv, const unsigned short* __restrict__ Ha,
    const float* __restrict__ V_W2, const float* __restrict__ V_b2,
    const float* __restrict__ A_W2, const float* __restrict__ A_b2,
    float* __restrict__ Vd, float* __restrict__ Ad)
{
  __shared__ __align__(16) unsigned short Xs[2][64][LDSK];   // 10 KB
  __shared__ __align__(16) unsigned short Ws[2][DD][LDSK];   // 20 KB
  const int n    = blockIdx.x;
  const int r0   = blockIdx.y * 64;
  const int path = blockIdx.z;
  const unsigned short* Hin = path ? Ha : Hv;
  const float* W2 = path ? A_W2 : V_W2;
  const float* b2 = path ? A_b2 : V_b2;
  float* Out = path ? Ad : Vd;

  const int tid  = threadIdx.x;
  const int lane = tid & 31;
  const int w    = tid >> 5;
  const int wm   = w >> 2, wn = w & 3;   // wave tile 32x32
  const int l15  = lane & 15;
  const int halfsel = lane >> 4;

  // staging thread mapping
  const int hb   = tid >> 2;            // H: row 0..63
  const int hseg = (tid & 3) << 3;      // H: 8-short segment
  const int wog   = (tid & 31) << 2;    // W2: 4 output columns
  const int wkbas = (tid >> 5) << 2;    // W2: 4 consecutive k

  u32x4 hr;
  f32x4 wr[4];

  auto stage_load = [&](int k0) {
    hr = *(const u32x4*)&Hin[((size_t)n*BB + r0 + hb)*HH + k0 + hseg];
    #pragma unroll
    for (int j = 0; j < 4; ++j)
      wr[j] = *(const f32x4*)&W2[((size_t)n*HH + (k0 + wkbas + j))*DD + wog];
  };
  auto stage_store = [&](int buf) {
    *(u32x4*)&Xs[buf][hb][hseg] = hr;
    #pragma unroll
    for (int c = 0; c < 4; ++c) {
      u32x2 p;
      p.x = pack_bf16(wr[0][c], wr[1][c]);
      p.y = pack_bf16(wr[2][c], wr[3][c]);
      *(u32x2*)&Ws[buf][wog + c][wkbas] = p;
    }
  };

  v8f acc[2][2];
  #pragma unroll
  for (int a = 0; a < 2; ++a)
    #pragma unroll
    for (int c = 0; c < 2; ++c) {
      v8f z = {0.f,0.f,0.f,0.f,0.f,0.f,0.f,0.f};
      acc[a][c] = z;
    }

  stage_load(0);
  stage_store(0);
  __syncthreads();
  for (int kc = 0; kc < 8; ++kc) {
    const int cur = kc & 1;
    const bool more = (kc + 1 < 8);
    if (more) stage_load((kc + 1) << 5);
    v16bf af[2], bfr[2];
    #pragma unroll
    for (int tm = 0; tm < 2; ++tm)
      af[tm] = frag_a(&Xs[cur][wm*32 + tm*16 + l15][0], lane);
    #pragma unroll
    for (int tn = 0; tn < 2; ++tn)
      bfr[tn] = frag_b(&Ws[cur][wn*32 + tn*16 + l15][0], lane);
    #pragma unroll
    for (int tm = 0; tm < 2; ++tm)
      #pragma unroll
      for (int tn = 0; tn < 2; ++tn)
        acc[tm][tn] = __builtin_amdgcn_wmma_f32_16x16x32_bf16(
            false, af[tm], false, bfr[tn], (short)0, acc[tm][tn], false, false);
    if (more) stage_store(cur ^ 1);
    __syncthreads();
  }

  #pragma unroll
  for (int tn = 0; tn < 2; ++tn) {
    int col = wn*32 + tn*16 + l15;
    float bias = b2[(size_t)n*DD + col];
    #pragma unroll
    for (int tm = 0; tm < 2; ++tm)
      #pragma unroll
      for (int v = 0; v < 8; ++v) {
        int row = wm*32 + tm*16 + v + (halfsel ? 8 : 0);
        Out[((size_t)n*BB + r0 + row)*DD + col] = acc[tm][tn][v] + bias;
      }
  }
}

// ---------------------------------------------------------------------------
// Kernel 3: chi head. grid = (n=64, mb=4); wave per b, lanes cover d via f32x4.
// (d remap to (lane, component) is legal: only a mean over d is taken.)
// ---------------------------------------------------------------------------
__device__ __forceinline__ f32x4 min4(f32x4 a, f32x4 b) {
  f32x4 r;
  r.x = fminf(a.x, b.x); r.y = fminf(a.y, b.y);
  r.z = fminf(a.z, b.z); r.w = fminf(a.w, b.w);
  return r;
}

__global__ __launch_bounds__(256) void chi_kernel(
    const float* __restrict__ Vd, const float* __restrict__ Ad,  // (N,B,D)
    const int*   __restrict__ edges,                             // (N,2,K+1)
    const float* __restrict__ m1,                                // (N,3,K)
    const float* __restrict__ m2,                                // (N,3,K,K)
    float* __restrict__ out)
{
  __shared__ float s_w1[KK];
  __shared__ float s_M2[KK][KK];
  __shared__ int   s_neigh[KK];
  __shared__ int   s_center;
  const int n    = blockIdx.x;
  const int b0   = blockIdx.y * 32;
  const int tid  = threadIdx.x;
  const int lane = tid & 31;
  const int w    = tid >> 5;

  if (tid == 0) s_center = edges[n*2*(KK+1)];
  if (tid < KK) {
    s_neigh[tid] = edges[n*2*(KK+1) + 1 + tid];
    float s = 0.f;
    for (int h = 0; h < 3; ++h) s += m1[((size_t)n*3 + h)*KK + tid];
    s_w1[tid] = s;
  }
  if (tid < KK*KK) {
    int i = tid >> 3, j = tid & 7;
    float s = 0.f;
    if (j > i)
      for (int h = 0; h < 3; ++h) s += m2[(((size_t)n*3 + h)*KK + i)*KK + j];
    s_M2[i][j] = s;   // triu(k=1) mask folded in
  }
  __syncthreads();

  const int doff = lane << 2;   // 4 consecutive d per lane
  for (int t = 0; t < 4; ++t) {
    const int b = b0 + t*8 + w;
    f32x4 xv[KK], xq[KK];
    #pragma unroll
    for (int k = 0; k < KK; ++k) {
      int np = s_neigh[k];
      f32x4 v = *(const f32x4*)&Vd[((size_t)np*BB + b)*DD + doff];
      f32x4 a = *(const f32x4*)&Ad[((size_t)np*BB + b)*DD + doff];
      xv[k] = v;
      xq[k] = v + a;
    }
    f32x4 aq = {0.f,0.f,0.f,0.f}, av = {0.f,0.f,0.f,0.f};
    #pragma unroll
    for (int k = 0; k < KK; ++k) {
      float wk = s_w1[k];
      aq += wk * xq[k];
      av += wk * xv[k];
    }
    #pragma unroll
    for (int i = 0; i < KK; ++i)
      #pragma unroll
      for (int j = i + 1; j < KK; ++j) {
        float m = s_M2[i][j];
        aq += m * min4(xq[i], xq[j]);
        av += m * min4(xv[i], xv[j]);
      }
    int cn = s_center;
    f32x4 cv = *(const f32x4*)&Vd[((size_t)cn*BB + b)*DD + doff];
    f32x4 ca = *(const f32x4*)&Ad[((size_t)cn*BB + b)*DD + doff];
    const float inv3 = 1.0f/3.0f;
    f32x4 tq = aq * inv3 + (cv + ca);
    f32x4 tv = av * inv3 + cv;
    float sq = tq.x + tq.y + tq.z + tq.w;
    float sv = tv.x + tv.y + tv.z + tv.w;
    #pragma unroll
    for (int off = 1; off < 32; off <<= 1) {
      sq += __shfl_xor(sq, off, 32);
      sv += __shfl_xor(sv, off, 32);
    }
    if (lane == 0) {
      out[(size_t)b*NB + n]                 = sq * (1.f/128.f);  // chi_q
      out[(size_t)BB*NB + (size_t)b*NB + n] = sv * (1.f/128.f);  // chi_v
    }
  }
}

// ---------------------------------------------------------------------------
extern "C" void kernel_launch(void* const* d_in, const int* in_sizes, int n_in,
                              void* d_out, int out_size, void* d_ws, size_t ws_size,
                              hipStream_t stream) {
  const float* obs   = (const float*)d_in[0];
  const float* act   = (const float*)d_in[1];
  const int*   edges = (const int*)  d_in[2];
  const float* V_W1  = (const float*)d_in[3];
  const float* V_b1  = (const float*)d_in[4];
  const float* V_g1  = (const float*)d_in[5];
  const float* V_be1 = (const float*)d_in[6];
  const float* V_W2  = (const float*)d_in[7];
  const float* V_b2  = (const float*)d_in[8];
  const float* A_W1  = (const float*)d_in[9];
  const float* A_b1  = (const float*)d_in[10];
  const float* A_g1  = (const float*)d_in[11];
  const float* A_be1 = (const float*)d_in[12];
  const float* A_W2  = (const float*)d_in[13];
  const float* A_b2  = (const float*)d_in[14];
  const float* m1    = (const float*)d_in[15];
  const float* m2    = (const float*)d_in[16];
  float* out = (float*)d_out;

  // workspace layout (needs 16 MB):
  char* ws = (char*)d_ws;
  unsigned short* Hv = (unsigned short*)(ws);                         // bf16 (N,B,H)
  unsigned short* Ha = (unsigned short*)(ws + (size_t)4*1024*1024);   // bf16 (N,B,H)
  float* Vd = (float*)(ws + (size_t) 8*1024*1024);                    // f32 (N,B,D)
  float* Ad = (float*)(ws + (size_t)12*1024*1024);                    // f32 (N,B,D)

  mlp1_ln_relu<<<dim3(NB, 2, 2), 256, 0, stream>>>(
      obs, act, V_W1, V_b1, V_g1, V_be1, A_W1, A_b1, A_g1, A_be1, Hv, Ha);
  mlp2_kernel<<<dim3(NB, 2, 2), 256, 0, stream>>>(
      Hv, Ha, V_W2, V_b2, A_W2, A_b2, Vd, Ad);
  chi_kernel<<<dim3(NB, 4), 256, 0, stream>>>(Vd, Ad, edges, m1, m2, out);
}
